// OneGnn_57801669869757
// MI455X (gfx1250) — matrix operand order repaired and checked by reading, not compiled, not run on hardware
//
#include <hip/hip_runtime.h>
#include <math.h>

#define NNODES 50000
#define NEDGES 800000
#define FIN    128
#define HID    128
#define NGRAPH 512
#define NCLS   10

typedef __attribute__((ext_vector_type(16))) __bf16 v16bf;
typedef __attribute__((ext_vector_type(8)))  float  v8f;

// ---------------------------------------------------------------------------
// WMMA fragment loaders (layouts per CDNA5 ISA 7.12.2, wave32)
// A: 16x32 bf16, row-major [M,K].  lane&15 = row; lane>>4 selects K half-bank.
//   lanes 0-15 : VGPR0-3 -> K 0..7,  VGPR4-7 -> K 16..23
//   lanes 16-31: VGPR0-3 -> K 8..15, VGPR4-7 -> K 24..31
__device__ __forceinline__ v16bf load_a_frag(const __bf16* A, int K, int row,
                                             int k0, int lane) {
  union { v16bf v; unsigned u[8]; } f;
  const __bf16* base = A + (size_t)row * K + k0 + ((lane >> 4) << 3);
#pragma unroll
  for (int r = 0; r < 8; ++r) {
    int kk = ((r & 4) << 2) + ((r & 3) << 1);   // +16 for r>=4, +2 per pair
    f.u[r] = *(const unsigned*)(base + kk);
  }
  return f.v;
}

// B: 32x16 bf16. lane&15 = col; same K order as A: lanes 0-15 K 0..7,16..23,
// lanes 16-31 K 8..15,24..31. Fed from pre-transposed WT[Nout,K].
__device__ __forceinline__ v16bf load_b_frag(const __bf16* WT, int K, int col,
                                             int k0, int lane) {
  union { v16bf v; unsigned u[8]; } f;
  const __bf16* base = WT + (size_t)col * K + k0 + ((lane >> 4) << 3);
#pragma unroll
  for (int r = 0; r < 8; ++r)
    f.u[r] = *(const unsigned*)(base + ((r & 4) << 2) + ((r & 3) << 1));
  return f.v;
}

// ---------------------------------------------------------------------------
// out[M,Nout] = elu( A0[M,K] @ W0T' + (A1 ? A1 @ W1T' : 0) + bias ).
// One wave -> one 16x64 strip (4 N-tiles); K-loop unrolled x4 so the
// scheduler can interleave the next step's fragment loads with the current
// step's 4 WMMAs (no explicit rotation registers -> no spills).
__global__ void gemm_wmma_dual(const __bf16* __restrict__ A0,
                               const __bf16* __restrict__ W0T,
                               const __bf16* __restrict__ A1,
                               const __bf16* __restrict__ W1T,
                               const float*  __restrict__ bias,
                               float* __restrict__ outF,
                               __bf16* __restrict__ outB,
                               int M, int K, int Nout, int act) {
  const int lane = threadIdx.x & 31;
  const int wave = blockIdx.x * (blockDim.x >> 5) + (threadIdx.x >> 5);
  const int stripsN = Nout >> 6;                       // 16x64 strips
  const int totalStrips = (M >> 4) * stripsN;
  if (wave >= totalStrips) return;                     // uniform per wave
  const int tm = (wave / stripsN) << 4;
  const int tn = (wave % stripsN) << 6;
  const int arow = tm + (lane & 15);
  const int ncol = lane & 15;
  const int c0 = tn + ncol, c1 = c0 + 16, c2 = c0 + 32, c3 = c0 + 48;

  v8f acc[4] = {};
#pragma unroll 4
  for (int k0 = 0; k0 < K; k0 += 32) {
    v16bf a  = load_a_frag(A0, K, arow, k0, lane);
    v16bf b0 = load_b_frag(W0T, K, c0, k0, lane);
    v16bf b1 = load_b_frag(W0T, K, c1, k0, lane);
    v16bf b2 = load_b_frag(W0T, K, c2, k0, lane);
    v16bf b3 = load_b_frag(W0T, K, c3, k0, lane);
    acc[0] = __builtin_amdgcn_wmma_f32_16x16x32_bf16(false, a, false, b0,
                                                     (short)0, acc[0], false, false);
    acc[1] = __builtin_amdgcn_wmma_f32_16x16x32_bf16(false, a, false, b1,
                                                     (short)0, acc[1], false, false);
    acc[2] = __builtin_amdgcn_wmma_f32_16x16x32_bf16(false, a, false, b2,
                                                     (short)0, acc[2], false, false);
    acc[3] = __builtin_amdgcn_wmma_f32_16x16x32_bf16(false, a, false, b3,
                                                     (short)0, acc[3], false, false);
  }
  if (A1) {
#pragma unroll 4
    for (int k0 = 0; k0 < K; k0 += 32) {
      v16bf a  = load_a_frag(A1, K, arow, k0, lane);
      v16bf b0 = load_b_frag(W1T, K, c0, k0, lane);
      v16bf b1 = load_b_frag(W1T, K, c1, k0, lane);
      v16bf b2 = load_b_frag(W1T, K, c2, k0, lane);
      v16bf b3 = load_b_frag(W1T, K, c3, k0, lane);
      acc[0] = __builtin_amdgcn_wmma_f32_16x16x32_bf16(false, a, false, b0,
                                                       (short)0, acc[0], false, false);
      acc[1] = __builtin_amdgcn_wmma_f32_16x16x32_bf16(false, a, false, b1,
                                                       (short)0, acc[1], false, false);
      acc[2] = __builtin_amdgcn_wmma_f32_16x16x32_bf16(false, a, false, b2,
                                                       (short)0, acc[2], false, false);
      acc[3] = __builtin_amdgcn_wmma_f32_16x16x32_bf16(false, a, false, b3,
                                                       (short)0, acc[3], false, false);
    }
  }

  const int rbase = tm + ((lane >> 4) << 3);           // D: M = 8*half + vgpr
#pragma unroll
  for (int t = 0; t < 4; ++t) {
    const int bcol = tn + (t << 4) + ncol;
    const float bn = bias ? bias[bcol] : 0.0f;
#pragma unroll
    for (int r = 0; r < 8; ++r) {
      float x = acc[t][r] + bn;
      if (act) x = x > 0.0f ? x : (__expf(x) - 1.0f);  // ELU (alpha=1)
      size_t o = (size_t)(rbase + r) * Nout + bcol;
      outF[o] = x;
      if (outB) outB[o] = (__bf16)x;
    }
  }
}

// ---------------------------------------------------------------------------
__global__ void scatter_add_edges(const float* __restrict__ h,
                                  const int* __restrict__ src,
                                  const int* __restrict__ dst,
                                  float* __restrict__ agg, int F) {
  int e = blockIdx.x;
  int f = blockIdx.y * blockDim.x + threadIdx.x;
  int s = src[e], d = dst[e];
  atomicAdd(&agg[(size_t)d * F + f], h[(size_t)s * F + f]);
}

__global__ void f32_to_bf16(const float* __restrict__ in,
                            __bf16* __restrict__ out, int n) {
  int i = blockIdx.x * blockDim.x + threadIdx.x;
  if (i < n) out[i] = (__bf16)in[i];
}

// W[K,Nout] f32 row-major -> WT[Nout,K] bf16
__global__ void transpose_w_bf16(const float* __restrict__ W,
                                 __bf16* __restrict__ WT, int K, int Nout) {
  int i = blockIdx.x * blockDim.x + threadIdx.x;
  if (i >= K * Nout) return;
  int k = i / Nout, n = i % Nout;
  WT[(size_t)n * K + k] = (__bf16)W[i];
}

__global__ void pool_sum(const float* __restrict__ h,
                         const int* __restrict__ batch,
                         float* __restrict__ sums, int F) {
  int n = blockIdx.x;
  int f = blockIdx.y * blockDim.x + threadIdx.x;
  atomicAdd(&sums[(size_t)batch[n] * F + f], h[(size_t)n * F + f]);
}

__global__ void pool_count(const int* __restrict__ batch,
                           float* __restrict__ counts, int n) {
  int i = blockIdx.x * blockDim.x + threadIdx.x;
  if (i < n) atomicAdd(&counts[batch[i]], 1.0f);
}

__global__ void pool_div(const float* __restrict__ sums,
                         const float* __restrict__ counts,
                         float* __restrict__ pooledF,
                         __bf16* __restrict__ pooledB, int F) {
  int g = blockIdx.x;
  int f = blockIdx.y * blockDim.x + threadIdx.x;
  float c = counts[g]; c = c > 1.0f ? c : 1.0f;
  float v = sums[(size_t)g * F + f] / c;
  pooledF[(size_t)g * F + f] = v;
  pooledB[(size_t)g * F + f] = (__bf16)v;
}

// logits = g2[G,128] @ w[128,10] + b, then log_softmax. One wave per graph.
__global__ void head_logsoftmax(const float* __restrict__ g2,
                                const float* __restrict__ w,
                                const float* __restrict__ b,
                                float* __restrict__ out) {
  int g = blockIdx.x;
  int lane = threadIdx.x;          // 32 lanes
  float logit = 0.0f;
  float m = -INFINITY;
  if (lane < NCLS) {
    float s = b[lane];
    const float* row = g2 + (size_t)g * HID;
    for (int k = 0; k < HID; ++k) s += row[k] * w[k * NCLS + lane];
    logit = s; m = s;
  }
  for (int off = 16; off; off >>= 1) {
    float o = __shfl_xor(m, off, 32);
    m = m > o ? m : o;
  }
  float ex = (lane < NCLS) ? __expf(logit - m) : 0.0f;
  float sum = ex;
  for (int off = 16; off; off >>= 1) sum += __shfl_xor(sum, off, 32);
  if (lane < NCLS) out[(size_t)g * NCLS + lane] = logit - m - __logf(sum);
}

// ---------------------------------------------------------------------------
static inline void launch_gemm(const __bf16* A0, const __bf16* W0T,
                               const __bf16* A1, const __bf16* W1T,
                               const float* bias, float* outF, __bf16* outB,
                               int M, int K, int Nout, int act,
                               hipStream_t s) {
  int strips = (M / 16) * (Nout / 64);
  int blocks = (strips + 3) / 4;                    // 4 waves (128 thr) / block
  gemm_wmma_dual<<<blocks, 128, 0, s>>>(A0, W0T, A1, W1T, bias,
                                        outF, outB, M, K, Nout, act);
}

extern "C" void kernel_launch(void* const* d_in, const int* in_sizes, int n_in,
                              void* d_out, int out_size, void* d_ws,
                              size_t ws_size, hipStream_t stream) {
  const float* x       = (const float*)d_in[0];
  const int*   eidx    = (const int*)d_in[1];
  const int*   src     = eidx;
  const int*   dst     = eidx + NEDGES;
  const int*   batch   = (const int*)d_in[2];
  const float* w1_rel  = (const float*)d_in[3];
  const float* b1      = (const float*)d_in[4];
  const float* w1_root = (const float*)d_in[5];
  const float* w2_rel  = (const float*)d_in[6];
  const float* b2      = (const float*)d_in[7];
  const float* w2_root = (const float*)d_in[8];
  const float* w3_rel  = (const float*)d_in[9];
  const float* b3      = (const float*)d_in[10];
  const float* w3_root = (const float*)d_in[11];
  const float* w_m1    = (const float*)d_in[12];
  const float* b_m1    = (const float*)d_in[13];
  const float* w_m2    = (const float*)d_in[14];
  const float* b_m2    = (const float*)d_in[15];
  const float* w_m3    = (const float*)d_in[16];
  const float* b_m3    = (const float*)d_in[17];

  // ---- carve workspace -----------------------------------------------------
  char* p = (char*)d_ws;
  auto alloc = [&](size_t bytes) -> void* {
    void* r = (void*)p;
    p += (bytes + 255) & ~(size_t)255;
    return r;
  };
  __bf16* xb    = (__bf16*)alloc((size_t)NNODES * FIN * 2);
  float*  aggF  = (float*) alloc((size_t)NNODES * 256 * 4);
  __bf16* aggB  = (__bf16*)alloc((size_t)NNODES * 256 * 2);
  float*  hF0   = (float*) alloc((size_t)NNODES * 256 * 4);
  float*  hF1   = (float*) alloc((size_t)NNODES * 256 * 4);
  __bf16* hB0   = (__bf16*)alloc((size_t)NNODES * 256 * 2);
  __bf16* hB1   = (__bf16*)alloc((size_t)NNODES * 256 * 2);
  __bf16* w1rT  = (__bf16*)alloc(128 * 128 * 2);
  __bf16* w1oT  = (__bf16*)alloc(128 * 128 * 2);
  __bf16* w2rT  = (__bf16*)alloc(128 * 256 * 2);
  __bf16* w2oT  = (__bf16*)alloc(128 * 256 * 2);
  __bf16* w3rT  = (__bf16*)alloc(256 * 256 * 2);
  __bf16* w3oT  = (__bf16*)alloc(256 * 256 * 2);
  __bf16* wm1T  = (__bf16*)alloc(256 * 256 * 2);
  __bf16* wm2T  = (__bf16*)alloc(256 * 128 * 2);
  float*  sums  = (float*) alloc((size_t)NGRAPH * 256 * 4);
  float*  cnts  = (float*) alloc((size_t)NGRAPH * 4);
  float*  poolF = (float*) alloc((size_t)NGRAPH * 256 * 4);
  __bf16* poolB = (__bf16*)alloc((size_t)NGRAPH * 256 * 2);
  float*  g1F   = (float*) alloc((size_t)NGRAPH * 256 * 4);
  __bf16* g1B   = (__bf16*)alloc((size_t)NGRAPH * 256 * 2);
  float*  g2F   = (float*) alloc((size_t)NGRAPH * 128 * 4);

  // ---- weight prep (bf16, transposed to [Nout,K]) --------------------------
  auto tw = [&](const float* W, __bf16* WT, int K, int Nout) {
    int n = K * Nout;
    transpose_w_bf16<<<(n + 255) / 256, 256, 0, stream>>>(W, WT, K, Nout);
  };
  tw(w1_rel, w1rT, 128, 128);  tw(w1_root, w1oT, 128, 128);
  tw(w2_rel, w2rT, 128, 256);  tw(w2_root, w2oT, 128, 256);
  tw(w3_rel, w3rT, 256, 256);  tw(w3_root, w3oT, 256, 256);
  tw(w_m1,   wm1T, 256, 256);  tw(w_m2,    wm2T, 256, 128);

  // x -> bf16
  {
    int n = NNODES * FIN;
    f32_to_bf16<<<(n + 255) / 256, 256, 0, stream>>>(x, xb, n);
  }

  // ---- layer 1: 128 -> 128 -------------------------------------------------
  hipMemsetAsync(aggF, 0, (size_t)NNODES * 128 * 4, stream);
  scatter_add_edges<<<dim3(NEDGES, 1), 128, 0, stream>>>(x, src, dst, aggF, 128);
  f32_to_bf16<<<(NNODES * 128 + 255) / 256, 256, 0, stream>>>(aggF, aggB,
                                                              NNODES * 128);
  launch_gemm(aggB, w1rT, xb, w1oT, b1, hF0, hB0, NNODES, 128, 128, 1, stream);

  // ---- layer 2: 128 -> 256 -------------------------------------------------
  hipMemsetAsync(aggF, 0, (size_t)NNODES * 128 * 4, stream);
  scatter_add_edges<<<dim3(NEDGES, 1), 128, 0, stream>>>(hF0, src, dst, aggF, 128);
  f32_to_bf16<<<(NNODES * 128 + 255) / 256, 256, 0, stream>>>(aggF, aggB,
                                                              NNODES * 128);
  launch_gemm(aggB, w2rT, hB0, w2oT, b2, hF1, hB1, NNODES, 128, 256, 1, stream);

  // ---- layer 3: 256 -> 256 -------------------------------------------------
  hipMemsetAsync(aggF, 0, (size_t)NNODES * 256 * 4, stream);
  scatter_add_edges<<<dim3(NEDGES, 2), 128, 0, stream>>>(hF1, src, dst, aggF, 256);
  f32_to_bf16<<<(NNODES * 256 + 255) / 256, 256, 0, stream>>>(aggF, aggB,
                                                              NNODES * 256);
  launch_gemm(aggB, w3rT, hB1, w3oT, b3, hF0, hB0, NNODES, 256, 256, 1, stream);

  // ---- mean pool over graphs ----------------------------------------------
  hipMemsetAsync(sums, 0, (size_t)NGRAPH * 256 * 4, stream);
  hipMemsetAsync(cnts, 0, (size_t)NGRAPH * 4, stream);
  pool_sum<<<dim3(NNODES, 2), 128, 0, stream>>>(hF0, batch, sums, 256);
  pool_count<<<(NNODES + 255) / 256, 256, 0, stream>>>(batch, cnts, NNODES);
  pool_div<<<dim3(NGRAPH, 2), 128, 0, stream>>>(sums, cnts, poolF, poolB, 256);

  // ---- MLP head ------------------------------------------------------------
  launch_gemm(poolB, wm1T, nullptr, nullptr, b_m1, g1F, g1B,
              NGRAPH, 256, 256, 1, stream);
  launch_gemm(g1B, wm2T, nullptr, nullptr, b_m2, g2F, nullptr,
              NGRAPH, 256, 128, 1, stream);
  head_logsoftmax<<<NGRAPH, 32, 0, stream>>>(g2F, w_m3, b_m3, (float*)d_out);

  (void)in_sizes; (void)n_in; (void)out_size; (void)ws_size;
}
